// Attention_77455440216322
// MI455X (gfx1250) — matrix operand order repaired and checked by reading, not compiled, not run on hardware
//
#include <hip/hip_runtime.h>
#include <hip/hip_bf16.h>

typedef __bf16 bf16;
typedef __attribute__((ext_vector_type(16))) __bf16 v16bf;
typedef __attribute__((ext_vector_type(8)))  float   v8f;

#define BB 4
#define NN 2048
#define DD 1024
#define HH 16
#define HD 64
#define MROWS (BB*NN)      // 8192
#define QKVN  (3*DD)       // 3072

union V16 {
    v16bf v;
    uint4 u[2];
    __bf16 e[16];
};

// ---------------------------------------------------------------------------
// Fragment loaders (CDNA5 ISA 16-bit WMMA VGPR layouts, wave32)
// ---------------------------------------------------------------------------

// A matrix 16x32 (MxK), row-major source, ld = row stride (elements).
// lane<16: row=lane, K=0..7 (v0-3), K=16..23 (v4-7); lane>=16: K=8..15, 24..31.
__device__ __forceinline__ v16bf load_frag_a(const bf16* __restrict__ base, int ld) {
    int lane = threadIdx.x & 31;
    int row  = lane & 15;
    int kb   = (lane >> 4) * 8;
    const bf16* p = base + (size_t)row * ld + kb;
    V16 r;
    r.u[0] = *(const uint4*)(p);
    r.u[1] = *(const uint4*)(p + 16);
    return r.v;
}

// B matrix 32x16 (KxN). Source "Bt" layout: Bt[n][k], k contiguous, ld = K stride.
// lane = column n (mod 16); lanes 0-15 hold K=0..7,16..23, lanes 16-31 K=8..15,24..31.
__device__ __forceinline__ v16bf load_frag_b(const bf16* __restrict__ base, int ld) {
    int lane = threadIdx.x & 31;
    int col  = lane & 15;
    int kb   = (lane >> 4) * 8;
    const bf16* p = base + (size_t)col * ld + kb;
    V16 r;
    r.u[0] = *(const uint4*)(p);
    r.u[1] = *(const uint4*)(p + 16);
    return r.v;
}

__device__ __forceinline__ v8f wmma_bf16(v16bf a, v16bf b, v8f c) {
    return __builtin_amdgcn_wmma_f32_16x16x32_bf16(
        false, a, false, b, (short)0, c, false, false);
}

// ---------------------------------------------------------------------------
// 1) LayerNorm -> bf16
// ---------------------------------------------------------------------------
__global__ void ln_kernel(const float* __restrict__ x, const float* __restrict__ g,
                          const float* __restrict__ be, bf16* __restrict__ xn) {
    int row = blockIdx.x;
    const float* xr = x + (size_t)row * DD;
    float s = 0.f, s2 = 0.f;
    for (int i = threadIdx.x; i < DD; i += 256) {
        float v = xr[i];
        s += v; s2 += v * v;
    }
    for (int o = 16; o > 0; o >>= 1) {
        s  += __shfl_xor(s,  o, 32);
        s2 += __shfl_xor(s2, o, 32);
    }
    __shared__ float red[2][8];
    int w = threadIdx.x >> 5;
    if ((threadIdx.x & 31) == 0) { red[0][w] = s; red[1][w] = s2; }
    __syncthreads();
    float ts = 0.f, ts2 = 0.f;
    for (int i = 0; i < 8; i++) { ts += red[0][i]; ts2 += red[1][i]; }
    float mu   = ts * (1.f / DD);
    float var  = ts2 * (1.f / DD) - mu * mu;
    float rstd = rsqrtf(var + 1e-5f);
    bf16* xo = xn + (size_t)row * DD;
    for (int i = threadIdx.x; i < DD; i += 256)
        xo[i] = (bf16)((xr[i] - mu) * rstd * g[i] + be[i]);
}

// ---------------------------------------------------------------------------
// 2) LDS-tiled weight transpose + fp32->bf16: dst[n][k] = src[k][n]
// ---------------------------------------------------------------------------
__global__ void transpose_to_bf16(const float* __restrict__ src, bf16* __restrict__ dst,
                                  int K, int Ncol) {
    __shared__ float t[32][33];
    int n0 = blockIdx.x * 32;
    int k0 = blockIdx.y * 32;
    int tx = threadIdx.x & 31, ty = threadIdx.x >> 5;   // ty: 0..7
    #pragma unroll
    for (int i = 0; i < 4; i++) {
        int k = k0 + ty + i * 8;
        t[ty + i * 8][tx] = src[(size_t)k * Ncol + n0 + tx];
    }
    __syncthreads();
    #pragma unroll
    for (int i = 0; i < 4; i++) {
        int n = n0 + ty + i * 8;
        dst[(size_t)n * K + k0 + tx] = (bf16)t[tx][ty + i * 8];
    }
}

// ---------------------------------------------------------------------------
// 3/7) bf16 GEMM: C[MxN] = A[MxK] * Bt[NxK]^T (+bias), 64x64 wave tiles
// ---------------------------------------------------------------------------
template <typename OutT>
__global__ __launch_bounds__(128)
void gemm_bf16_kernel(const bf16* __restrict__ A, const bf16* __restrict__ Bt,
                      OutT* __restrict__ C, const float* __restrict__ bias,
                      int M, int Ncols, int K) {
    int lane   = threadIdx.x & 31;
    int wave   = blockIdx.x * (blockDim.x >> 5) + (threadIdx.x >> 5);
    int tilesN = Ncols >> 6;
    int tm = wave / tilesN, tn = wave % tilesN;
    if (tm >= (M >> 6)) return;
    int row0 = tm << 6, col0 = tn << 6;

    v8f acc[4][4] = {};
    for (int k = 0; k < K; k += 32) {
        v16bf a[4], b[4];
        #pragma unroll
        for (int i = 0; i < 4; i++)
            a[i] = load_frag_a(A + (size_t)(row0 + 16 * i) * K + k, K);
        #pragma unroll
        for (int j = 0; j < 4; j++)
            b[j] = load_frag_b(Bt + (size_t)(col0 + 16 * j) * K + k, K);
        if (k + 32 < K) {   // prefetch next k-slab (one line per lane)
            __builtin_prefetch(A  + (size_t)(row0 + lane)      * K + k + 32, 0, 3);
            __builtin_prefetch(A  + (size_t)(row0 + 32 + lane) * K + k + 32, 0, 3);
            __builtin_prefetch(Bt + (size_t)(col0 + lane)      * K + k + 32, 0, 3);
            __builtin_prefetch(Bt + (size_t)(col0 + 32 + lane) * K + k + 32, 0, 3);
        }
        #pragma unroll
        for (int i = 0; i < 4; i++)
            #pragma unroll
            for (int j = 0; j < 4; j++)
                acc[i][j] = wmma_bf16(a[i], b[j], acc[i][j]);
    }

    int cn = lane & 15, rg = (lane >> 4) << 3;   // C/D: row = r + 8*(lane/16)
    #pragma unroll
    for (int i = 0; i < 4; i++) {
        #pragma unroll
        for (int j = 0; j < 4; j++) {
            int col = col0 + 16 * j + cn;
            float bv = bias ? bias[col] : 0.f;
            #pragma unroll
            for (int r = 0; r < 8; r++) {
                int row = row0 + 16 * i + rg + r;
                C[(size_t)row * Ncols + col] = (OutT)(acc[i][j][r] + bv);
            }
        }
    }
}

// ---------------------------------------------------------------------------
// 4) RoPE on q,k (head reshape): qr,kr [b,h,n,d] bf16, q pre-scaled by HD^-0.5
// ---------------------------------------------------------------------------
__global__ void rope_kernel(const bf16* __restrict__ qkv, bf16* __restrict__ qr,
                            bf16* __restrict__ kr) {
    size_t idx = (size_t)blockIdx.x * 256 + threadIdx.x;
    if (idx >= (size_t)BB * HH * NN * 32) return;
    int d = (int)(idx & 31);
    size_t t = idx >> 5;
    int n = (int)(t & (NN - 1)); t >>= 11;
    int h = (int)(t & (HH - 1));
    int b = (int)(t >> 4);

    size_t rowoff = ((size_t)(b * NN + n)) * QKVN + h * HD;
    float q1 = (float)qkv[rowoff + d],      q2 = (float)qkv[rowoff + d + 32];
    float k1 = (float)qkv[rowoff + DD + d], k2 = (float)qkv[rowoff + DD + d + 32];

    float inv = __expf(-(float)(2 * d) * (9.210340371976184f / 64.f)); // 10000^(-2d/64)
    float ang = (float)n * inv;
    float sn = __sinf(ang), cs = __cosf(ang);
    const float sc = 0.125f; // HD^-0.5

    size_t ho = (((size_t)b * HH + h) * NN + n) * HD;
    qr[ho + d]      = (bf16)(sc * (q1 * cs - q2 * sn));
    qr[ho + d + 32] = (bf16)(sc * (q2 * cs + q1 * sn));
    kr[ho + d]      = (bf16)(k1 * cs - k2 * sn);
    kr[ho + d + 32] = (bf16)(k2 * cs + k1 * sn);
}

// ---------------------------------------------------------------------------
// 5) V transpose: vt[b,h,d,n] <- qkv v-slice, LDS-tiled, coalesced both sides
// ---------------------------------------------------------------------------
__global__ void vtrans_kernel(const bf16* __restrict__ qkv, bf16* __restrict__ vt) {
    __shared__ bf16 t[32][33];
    int n0 = blockIdx.x * 32;
    int d0 = blockIdx.y * 32;
    int bh = blockIdx.z;
    int b = bh >> 4, h = bh & (HH - 1);
    int tx = threadIdx.x & 31, ty = threadIdx.x >> 5;
    const bf16* src = qkv + (size_t)(b * NN) * QKVN + 2 * DD + h * HD + d0;
    #pragma unroll
    for (int i = 0; i < 4; i++) {
        int n = n0 + ty + i * 8;
        t[ty + i * 8][tx] = src[(size_t)n * QKVN + tx];   // t[n_local][d_local]
    }
    __syncthreads();
    bf16* dst = vt + ((size_t)bh * HD + d0) * NN + n0;
    #pragma unroll
    for (int i = 0; i < 4; i++) {
        int d = ty + i * 8;
        dst[(size_t)d * NN + tx] = t[tx][d];
    }
}

// ---------------------------------------------------------------------------
// 6) Flash attention, transposed scores (S^T = K·Q^T):
//    lane = query column -> softmax reductions in-lane (+1 xor-16 merge);
//    S^T C-layout IS the A-layout for the P·V WMMA (no LDS staging).
//    V fragments loaded before softmax so their latency hides under VALU work.
// ---------------------------------------------------------------------------
__global__ __launch_bounds__(128)
void attn_kernel(const bf16* __restrict__ qr, const bf16* __restrict__ kr,
                 const bf16* __restrict__ vt, bf16* __restrict__ attn) {
    int wid  = threadIdx.x >> 5;
    int lane = threadIdx.x & 31;
    int wave = blockIdx.x * 4 + wid;           // 0..8191
    int it = wave & 127;                        // query tile within (b,h)
    int bh = wave >> 7;                         // 0..63
    int h = bh & (HH - 1), b = bh >> 4;

    const bf16* Q  = qr + ((size_t)bh * NN + it * 16) * HD;
    const bf16* Kp = kr + (size_t)bh * NN * HD;
    const bf16* Vt = vt + (size_t)bh * HD * NN;

    // Q as B operand: B(k=d, n=query), d contiguous in [n][d] layout
    v16bf qb0 = load_frag_b(Q, HD);            // d = 0..31
    v16bf qb1 = load_frag_b(Q + 32, HD);       // d = 32..63

    v8f o[4] = {};                             // O: M=query, N=d (4 tiles of 16)
    float m = -1e30f, l = 0.f;                 // per query column (lane)

    int cn = lane & 15, rg = lane >> 4;

    for (int j = 0; j < NN; j += 32) {
        // K rows as A operand (issued first -> partial loadcnt waits for S WMMAs)
        v16bf ka00 = load_frag_a(Kp + (size_t)(j)      * HD,      HD);
        v16bf ka01 = load_frag_a(Kp + (size_t)(j)      * HD + 32, HD);
        v16bf ka10 = load_frag_a(Kp + (size_t)(j + 16) * HD,      HD);
        v16bf ka11 = load_frag_a(Kp + (size_t)(j + 16) * HD + 32, HD);

        // V fragments issued early: latency hidden under softmax VALU work
        v16bf vb[4];
        #pragma unroll
        for (int f = 0; f < 4; f++)
            vb[f] = load_frag_b(Vt + (size_t)(f * 16) * NN + j, NN);

        if (j + 32 < NN) {  // prefetch next j-block (one cacheline per lane)
            __builtin_prefetch(Kp + (size_t)(j + 32 + lane) * HD, 0, 3);
            __builtin_prefetch(Vt + (size_t)lane        * NN + j + 32, 0, 3);
            __builtin_prefetch(Vt + (size_t)(lane + 32) * NN + j + 32, 0, 3);
        }

        v8f s0 = {}, s1 = {};                  // S^T tiles: rows=keys, cols=queries
        s0 = wmma_bf16(ka00, qb0, s0);
        s0 = wmma_bf16(ka01, qb1, s0);
        s1 = wmma_bf16(ka10, qb0, s1);
        s1 = wmma_bf16(ka11, qb1, s1);

        // ---- online softmax: in-lane over 16 key values + one xor-16 merge
        float tmax = fmaxf(s0[0], s1[0]);
        #pragma unroll
        for (int r = 1; r < 8; r++)
            tmax = fmaxf(tmax, fmaxf(s0[r], s1[r]));
        tmax = fmaxf(tmax, __shfl_xor(tmax, 16, 32));

        float m_old = m;
        float mn = fmaxf(m_old, tmax);
        m = mn;
        float al = __expf(m_old - mn);

        V16 pa;                                 // P packed directly into A-layout
        float rs = 0.f;
        #pragma unroll
        for (int r = 0; r < 8; r++) {
            float e0 = __expf(s0[r] - mn);
            float e1 = __expf(s1[r] - mn);
            rs += e0 + e1;
            pa.e[r]     = (bf16)e0;             // S^T C-layout == A-layout halves
            pa.e[r + 8] = (bf16)e1;
        }
        rs += __shfl_xor(rs, 16, 32);
        l = l * al + rs;

        // rescale O only when some query's max moved (wave-uniform branch)
        if (__any(mn > m_old)) {
            #pragma unroll
            for (int r = 0; r < 8; r++) {
                float alr = __shfl(al, r + 8 * rg, 32);   // alpha of query row
                #pragma unroll
                for (int f = 0; f < 4; f++)
                    o[f][r] *= alr;
            }
        }

        // O += P x V
        #pragma unroll
        for (int f = 0; f < 4; f++)
            o[f] = wmma_bf16(pa.v, vb[f], o[f]);
    }

    // normalize rows by l (broadcast per query row) and store [b,n,h*64+d]
    size_t base = ((size_t)b * NN + it * 16) * (HH * HD) + h * HD;
    #pragma unroll
    for (int r = 0; r < 8; r++) {
        float lr = __shfl(l, r + 8 * rg, 32);
        float inv = 1.f / lr;
        int rowl = r + 8 * rg;
        #pragma unroll
        for (int f = 0; f < 4; f++) {
            int col = f * 16 + cn;
            attn[base + (size_t)rowl * (HH * HD) + col] = (bf16)(o[f][r] * inv);
        }
    }
}

// ---------------------------------------------------------------------------
// Launch
// ---------------------------------------------------------------------------
extern "C" void kernel_launch(void* const* d_in, const int* in_sizes, int n_in,
                              void* d_out, int out_size, void* d_ws, size_t ws_size,
                              hipStream_t stream) {
    const float* x    = (const float*)d_in[0];
    const float* g    = (const float*)d_in[1];
    const float* be   = (const float*)d_in[2];
    const float* wqkv = (const float*)d_in[3];
    const float* wout = (const float*)d_in[4];
    const float* bout = (const float*)d_in[5];
    float* out = (float*)d_out;

    const size_t SZ_XN    = (size_t)MROWS * DD * 2;
    const size_t SZ_WQKVT = (size_t)QKVN * DD * 2;
    const size_t SZ_WOUTT = (size_t)DD * DD * 2;
    const size_t SZ_QKV   = (size_t)MROWS * QKVN * 2;
    const size_t SZ_HEAD  = (size_t)BB * HH * NN * HD * 2;
    size_t off = 0;
    char* ws = (char*)d_ws;
    bf16* xn     = (bf16*)(ws + off); off += SZ_XN;
    bf16* wqkvT  = (bf16*)(ws + off); off += SZ_WQKVT;
    bf16* woutT  = (bf16*)(ws + off); off += SZ_WOUTT;
    bf16* qkv    = (bf16*)(ws + off); off += SZ_QKV;
    bf16* qrp    = (bf16*)(ws + off); off += SZ_HEAD;
    bf16* krp    = (bf16*)(ws + off); off += SZ_HEAD;
    bf16* vtp    = (bf16*)(ws + off); off += SZ_HEAD;
    bf16* attnb  = (bf16*)(ws + off); off += SZ_HEAD;
    if (off > ws_size) return;

    ln_kernel<<<MROWS, 256, 0, stream>>>(x, g, be, xn);

    transpose_to_bf16<<<dim3(QKVN / 32, DD / 32), 256, 0, stream>>>(wqkv, wqkvT, DD, QKVN);
    transpose_to_bf16<<<dim3(DD / 32,   DD / 32), 256, 0, stream>>>(wout, woutT, DD, DD);

    gemm_bf16_kernel<bf16><<<(MROWS / 64) * (QKVN / 64) / 4, 128, 0, stream>>>(
        xn, wqkvT, qkv, nullptr, MROWS, QKVN, DD);

    rope_kernel<<<(BB * HH * NN * 32) / 256, 256, 0, stream>>>(qkv, qrp, krp);
    vtrans_kernel<<<dim3(NN / 32, HD / 32, BB * HH), 256, 0, stream>>>(qkv, vtp);

    attn_kernel<<<(BB * HH * (NN / 16)) / 4, 128, 0, stream>>>(qrp, krp, vtp, attnb);

    gemm_bf16_kernel<float><<<(MROWS / 64) * (DD / 64) / 4, 128, 0, stream>>>(
        attnb, woutT, out, bout, MROWS, DD, DD);
}
